// FourierIntegralKernel1D_86895778333269
// MI455X (gfx1250) — matrix operand order repaired and checked by reading, not compiled, not run on hardware
//
#include <hip/hip_runtime.h>
#include <hip/hip_fp16.h>

typedef __attribute__((ext_vector_type(16))) _Float16 v16h;
typedef __attribute__((ext_vector_type(8)))  float    v8f;

#define N_LEN 8192
#define MODES 16
#define CIN   64
#define COUT  64
#define BATCH 32
#define MROWS (BATCH * CIN)        // 2048 rows for both GEMMs
#define KSEG  8
#define KSEG_LEN (N_LEN / KSEG)    // 1024

// Workspace layout (bytes):
//   T1T     : _Float16[32][8192]      @ 0        (forward twiddles, K-major: [2k]=cos, [2k+1]=-sin)
//   T2T     : _Float16[8192][32]      @ 524288   (inverse twiddles, N-major: [2k]=cos, [2k+1]=sin)
//   vf_part : float[8][32][2048]      @ 1048576  (per-K-segment partial spectra)
//   A2      : _Float16[2048][32]      @ 3145728  (packed irfft coefficients)
#define WS_T1T 0
#define WS_T2T 524288
#define WS_VFP 1048576
#define WS_A2  3145728

// ---------------------------------------------------------------------------
// Twiddle table generation (runs every launch; 128K sincos, negligible).
// ---------------------------------------------------------------------------
__global__ void fno_twiddle_init(_Float16* __restrict__ T1T,
                                 _Float16* __restrict__ T2T) {
    int idx = blockIdx.x * blockDim.x + threadIdx.x;   // 0 .. 8192*16-1
    int n = idx >> 4;
    int k = idx & 15;
    int p = (k * n) & (N_LEN - 1);                     // exact angle reduction
    float th = (float)p * (6.28318530717958647692f / (float)N_LEN);
    float s, c;
    __sincosf(th, &s, &c);
    T1T[(size_t)(2 * k)     * N_LEN + n] = (_Float16)c;
    T1T[(size_t)(2 * k + 1) * N_LEN + n] = (_Float16)(-s);
    T2T[(size_t)n * 32 + 2 * k]     = (_Float16)c;
    T2T[(size_t)n * 32 + 2 * k + 1] = (_Float16)s;
}

// ---------------------------------------------------------------------------
// Stage 1: truncated forward DFT as WMMA GEMM.
// M=2048 (b*Cin), K=8192 (time), Ncols=32 (16 modes x {re,im}).
// Each wave: one 16x16 tile of one K-segment (1024), 32 WMMAs.
// ---------------------------------------------------------------------------
__global__ void fno_dft_fwd(const float* __restrict__ v,
                            const _Float16* __restrict__ T1T,
                            float* __restrict__ vf_part) {
    int wave = blockIdx.x * (blockDim.x >> 5) + (threadIdx.x >> 5); // 0..2047
    int lane = threadIdx.x & 31;
    int kseg  = wave & 7;
    int ntile = (wave >> 3) & 1;
    int mtile = wave >> 4;          // 0..127
    int m0 = mtile << 4;
    int n0 = ntile << 4;
    int hi  = lane >> 4;            // 0 or 1 (upper half-wave)
    int klo = hi << 3;              // unused; A/B fragments use K sub-offset 16*hi
    int row = m0 + (lane & 15);     // A: lane -> M row
    int col = n0 + (lane & 15);     // B/C: lane -> N column

    const float*    arow = v   + (size_t)row * N_LEN;
    const _Float16* brow = T1T + (size_t)col * N_LEN;

    v8f acc = {};
    int kbeg = kseg * KSEG_LEN;
    for (int kk = kbeg; kk < kbeg + KSEG_LEN; kk += 32) {
        // A fragment (f32 -> f16): halves j -> K = kk + 16*hi + j
        const float4* p0 = (const float4*)(arow + kk + (hi << 4));
        const float4* p1 = (const float4*)(arow + kk + (hi << 4) + 8);
        float4 f0 = p0[0], f1 = p0[1], f2 = p1[0], f3 = p1[1];
        v16h a;
        a[0]  = (_Float16)f0.x; a[1]  = (_Float16)f0.y;
        a[2]  = (_Float16)f0.z; a[3]  = (_Float16)f0.w;
        a[4]  = (_Float16)f1.x; a[5]  = (_Float16)f1.y;
        a[6]  = (_Float16)f1.z; a[7]  = (_Float16)f1.w;
        a[8]  = (_Float16)f2.x; a[9]  = (_Float16)f2.y;
        a[10] = (_Float16)f2.z; a[11] = (_Float16)f2.w;
        a[12] = (_Float16)f3.x; a[13] = (_Float16)f3.y;
        a[14] = (_Float16)f3.z; a[15] = (_Float16)f3.w;
        // B fragment: 16 contiguous halves, halves j -> K = kk + 16*hi + j
        v16h b = *(const v16h*)(brow + kk + (hi << 4));
        acc = __builtin_amdgcn_wmma_f32_16x16x32_f16(
            false, a, false, b, (short)0, acc, false, false);
    }
    // C/D layout: VGPR r, lane -> (M = m0 + r + 8*hi, N = col)
    float* outp = vf_part + ((size_t)kseg * 32 + col) * MROWS + m0 + (hi << 3);
#pragma unroll
    for (int r = 0; r < 8; ++r) outp[r] = acc[r];
}

// ---------------------------------------------------------------------------
// Stage 2: complex mode mixing lv[b,o,k] = sum_i W[k,o,i] * vf[b,i,k],
// summing the 8 K-segment partials in fixed order (deterministic), then
// packing irfft coefficients: A2[row][2k]=w_k*re, A2[row][2k+1]=-w_k*im
// with w_0 = 1 (imag dropped, matching C2R irfft) and w_{k>=1} = 2.
// ---------------------------------------------------------------------------
__global__ void fno_mode_mix(const float* __restrict__ vf_part,
                             const float* __restrict__ w_real,
                             const float* __restrict__ w_imag,
                             _Float16* __restrict__ A2) {
    __shared__ float vre[CIN];
    __shared__ float vim[CIN];
    int k = blockIdx.x & 15;
    int b = blockIdx.x >> 4;
    int t = threadIdx.x;            // 0..63

    for (int base = 0; base < CIN; base += 32) {
        int i = base + (t >> 1);
        int c = t & 1;
        size_t off = (size_t)(2 * k + c) * MROWS + b * CIN + i;
        float s = 0.f;
#pragma unroll
        for (int seg = 0; seg < KSEG; ++seg)
            s += vf_part[(size_t)seg * (32 * MROWS) + off];
        if (c) vim[i] = s; else vre[i] = s;
    }
    __syncthreads();

    int o = t;
    const float* wr = w_real + ((size_t)k * COUT + o) * CIN;
    const float* wi = w_imag + ((size_t)k * COUT + o) * CIN;
    float re = 0.f, im = 0.f;
#pragma unroll 4
    for (int i = 0; i < CIN; ++i) {
        float vr = vre[i], vi = vim[i];
        re += wr[i] * vr - wi[i] * vi;
        im += wr[i] * vi + wi[i] * vr;
    }
    _Float16* outp = A2 + ((size_t)b * COUT + o) * 32 + 2 * k;
    if (k == 0) { outp[0] = (_Float16)re;         outp[1] = (_Float16)0.f; }
    else        { outp[0] = (_Float16)(2.f * re); outp[1] = (_Float16)(-2.f * im); }
}

// ---------------------------------------------------------------------------
// Stage 3: truncated inverse DFT as WMMA GEMM.
// M=2048 (b*Cout), K=32 (packed modes), Ncols=8192 (time).
// One WMMA per 16x16 output tile; 1/N scaling folded into the store.
// ---------------------------------------------------------------------------
__global__ void fno_idft(const _Float16* __restrict__ A2,
                         const _Float16* __restrict__ T2T,
                         float* __restrict__ out) {
    int wave = blockIdx.x * (blockDim.x >> 5) + (threadIdx.x >> 5); // 0..65535
    int lane = threadIdx.x & 31;
    int ntile = wave & 511;
    int mtile = wave >> 9;          // 0..127
    int m0 = mtile << 4;
    int n0 = ntile << 4;
    int hi  = lane >> 4;
    int klo = hi << 3;
    int row = m0 + (lane & 15);
    int n   = n0 + (lane & 15);

    const _Float16* ap = A2 + (size_t)row * 32;
    v16h a;
#pragma unroll
    for (int j = 0; j < 8; ++j) {
        a[j]     = ap[(hi << 4) + j];
        a[8 + j] = ap[(hi << 4) + 8 + j];
    }
    v16h b = *(const v16h*)(T2T + (size_t)n * 32 + (hi << 4));
    v8f c = {};
    c = __builtin_amdgcn_wmma_f32_16x16x32_f16(
        false, a, false, b, (short)0, c, false, false);

    float* op = out + (size_t)(m0 + (hi << 3)) * N_LEN + n;
#pragma unroll
    for (int r = 0; r < 8; ++r)
        op[(size_t)r * N_LEN] = c[r] * (1.0f / (float)N_LEN);
}

// ---------------------------------------------------------------------------
extern "C" void kernel_launch(void* const* d_in, const int* in_sizes, int n_in,
                              void* d_out, int out_size, void* d_ws, size_t ws_size,
                              hipStream_t stream) {
    const float* v      = (const float*)d_in[0];
    const float* w_real = (const float*)d_in[1];
    const float* w_imag = (const float*)d_in[2];
    float* out = (float*)d_out;

    char* ws = (char*)d_ws;
    _Float16* T1T     = (_Float16*)(ws + WS_T1T);
    _Float16* T2T     = (_Float16*)(ws + WS_T2T);
    float*    vf_part = (float*)   (ws + WS_VFP);
    _Float16* A2      = (_Float16*)(ws + WS_A2);

    // 1) Twiddle tables (131072 threads).
    fno_twiddle_init<<<(N_LEN * MODES) / 256, 256, 0, stream>>>(T1T, T2T);
    // 2) Forward truncated DFT: 2048 waves (128 M-tiles x 2 N-tiles x 8 K-segments).
    fno_dft_fwd<<<256, 256, 0, stream>>>(v, T1T, vf_part);
    // 3) Complex mode mixing + coefficient packing: 512 blocks x 64 threads.
    fno_mode_mix<<<MODES * BATCH, 64, 0, stream>>>(vf_part, w_real, w_imag, A2);
    // 4) Inverse truncated DFT: 65536 waves (128 x 512 tiles), 8 waves/block.
    fno_idft<<<8192, 256, 0, stream>>>(A2, T2T, out);
}
